// AnomalyDAEBase_1726576857664
// MI455X (gfx1250) — compile-verified, matrix operand order checked by reading, not run on hardware
//
#include <hip/hip_runtime.h>

#define N_NODES   10000
#define IN_DIM    128
#define EMB       64
#define HID       64
#define NEG_SLOPE 0.2f
#define NT        625   /* N_NODES / 16 */

typedef __attribute__((ext_vector_type(16))) _Float16 v16h;
typedef __attribute__((ext_vector_type(8)))  _Float16 v8h;
typedef __attribute__((ext_vector_type(8)))  float    v8f;

// ---------------- structure encoder: h = relu(x @ W_stru + b) ----------------
__global__ void k_h(const float* __restrict__ x, const float* __restrict__ W,
                    const float* __restrict__ b, float* __restrict__ h) {
  __shared__ float xs[IN_DIM];
  const int i = blockIdx.x, t = threadIdx.x;
  xs[t]      = x[(size_t)i * IN_DIM + t];
  xs[t + 64] = x[(size_t)i * IN_DIM + 64 + t];
  __syncthreads();
  float acc = b[t];
#pragma unroll 8
  for (int k = 0; k < IN_DIM; ++k) acc += xs[k] * W[k * EMB + t];
  h[(size_t)i * EMB + t] = fmaxf(acc, 0.f);
}

// ------------- hp = h @ W_gat ; a_src = hp.att_src ; a_dst = hp.att_dst -----
__global__ void k_hp(const float* __restrict__ h, const float* __restrict__ Wg,
                     const float* __restrict__ att_s, const float* __restrict__ att_d,
                     float* __restrict__ hp, float* __restrict__ a_src,
                     float* __restrict__ a_dst) {
  __shared__ float hs[EMB];
  __shared__ float red[HID];
  const int i = blockIdx.x, t = threadIdx.x;
  hs[t] = h[(size_t)i * EMB + t];
  __syncthreads();
  float acc = 0.f;
#pragma unroll 8
  for (int k = 0; k < EMB; ++k) acc += hs[k] * Wg[k * HID + t];
  hp[(size_t)i * HID + t] = acc;

  red[t] = acc * att_s[t];
  __syncthreads();
  for (int s = 32; s > 0; s >>= 1) { if (t < s) red[t] += red[t + s]; __syncthreads(); }
  if (t == 0) a_src[i] = red[0];
  __syncthreads();
  red[t] = acc * att_d[t];
  __syncthreads();
  for (int s = 32; s > 0; s >>= 1) { if (t < s) red[t] += red[t + s]; __syncthreads(); }
  if (t == 0) a_dst[i] = red[0];
}

// ---------------- CSR build over dst (real edges + self loops) ---------------
__global__ void k_zero_i32(int* __restrict__ p, int n) {
  int i = blockIdx.x * blockDim.x + threadIdx.x;
  if (i < n) p[i] = 0;
}

__global__ void k_count(const int* __restrict__ ei, int E, int* __restrict__ indeg) {
  int e = blockIdx.x * blockDim.x + threadIdx.x;
  if (e >= E + N_NODES) return;
  int dst = (e < E) ? ei[E + e] : (e - E);
  atomicAdd(&indeg[dst], 1);
}

__global__ void k_scan(const int* __restrict__ indeg, int* __restrict__ offsets,
                       int* __restrict__ cursor) {
  __shared__ int sdata[1024];
  const int t = threadIdx.x;
  const int CH = 10;                      // 1024*10 >= 10000
  const int base_i = t * CH;
  int sum = 0;
  for (int j = 0; j < CH; ++j) {
    int idx = base_i + j;
    if (idx < N_NODES) sum += indeg[idx];
  }
  sdata[t] = sum;
  __syncthreads();
  for (int off = 1; off < 1024; off <<= 1) {
    int v = sdata[t];
    if (t >= off) v += sdata[t - off];
    __syncthreads();
    sdata[t] = v;
    __syncthreads();
  }
  int run = (t == 0) ? 0 : sdata[t - 1];
  for (int j = 0; j < CH; ++j) {
    int idx = base_i + j;
    if (idx < N_NODES) {
      offsets[idx] = run;
      cursor[idx]  = run;
      run += indeg[idx];
    }
  }
  if (t == 0) offsets[N_NODES] = sdata[1023];
}

__global__ void k_fill(const int* __restrict__ ei, int E, int* __restrict__ cursor,
                       int* __restrict__ csr_src) {
  int e = blockIdx.x * blockDim.x + threadIdx.x;
  if (e >= E + N_NODES) return;
  int src, dst;
  if (e < E) { src = ei[e]; dst = ei[E + e]; }
  else       { src = e - E; dst = e - E; }
  int pos = atomicAdd(&cursor[dst], 1);
  csr_src[pos] = src;
}

// -------- GAT: segment softmax over incoming edges + weighted gather --------
__global__ void k_gat(const int* __restrict__ offsets, const int* __restrict__ csr_src,
                      const float* __restrict__ a_src, const float* __restrict__ a_dst,
                      const float* __restrict__ hp, const float* __restrict__ b_gat,
                      float* __restrict__ emb, _Float16* __restrict__ embh) {
  __shared__ float red[HID];
  const int i = blockIdx.x, t = threadIdx.x;
  const int beg = offsets[i], end = offsets[i + 1];
  const float adst = a_dst[i];

  // pass 1: segment max (order independent -> deterministic)
  float m = -3.4e38f;
  for (int e = beg + t; e < end; e += HID) {
    float v = a_src[csr_src[e]] + adst;
    v = (v > 0.f) ? v : NEG_SLOPE * v;
    m = fmaxf(m, v);
  }
  red[t] = m;
  __syncthreads();
  for (int s = 32; s > 0; s >>= 1) { if (t < s) red[t] = fmaxf(red[t], red[t + s]); __syncthreads(); }
  m = red[0];

  // pass 2: sequential per-node (thread = feature), no float atomics
  float acc = 0.f, den = 0.f;
  for (int e = beg; e < end; ++e) {
    int s = csr_src[e];
    float v = a_src[s] + adst;
    v = (v > 0.f) ? v : NEG_SLOPE * v;
    float w = __expf(v - m);
    den += w;
    acc += w * hp[(size_t)s * HID + t];
  }
  float o = acc / den + b_gat[t];
  emb[(size_t)i * HID + t]  = o;
  embh[(size_t)i * HID + t] = (_Float16)o;
}

// ------------- attribute encoder: xa = relu(x^T @ W_attr1 + b1) --------------
__global__ void k_attr1(const float* __restrict__ x, const float* __restrict__ W1,
                        const float* __restrict__ b1, float* __restrict__ xa) {
  const int i = blockIdx.x, t = threadIdx.x;   // i: input feature, t: emb dim
  float acc = b1[t];
#pragma unroll 4
  for (int k = 0; k < N_NODES; ++k)
    acc += x[(size_t)k * IN_DIM + i] * W1[(size_t)k * EMB + t];
  xa[i * EMB + t] = fmaxf(acc, 0.f);
}

// ---------------------- xa2 = xa @ W_attr2 + b2 ------------------------------
__global__ void k_attr2(const float* __restrict__ xa, const float* __restrict__ W2,
                        const float* __restrict__ b2, float* __restrict__ xa2) {
  __shared__ float xr[EMB];
  const int i = blockIdx.x, t = threadIdx.x;
  xr[t] = xa[i * EMB + t];
  __syncthreads();
  float acc = b2[t];
#pragma unroll 8
  for (int k = 0; k < EMB; ++k) acc += xr[k] * W2[k * HID + t];
  xa2[i * HID + t] = acc;
}

// ---------------------- x_ = emb @ xa2^T  -> d_out[0 : N*IN] -----------------
__global__ void k_xrec(const float* __restrict__ emb, const float* __restrict__ xa2,
                       float* __restrict__ out) {
  __shared__ float er[HID];
  const int i = blockIdx.x, t = threadIdx.x;   // t in [0,128)
  if (t < HID) er[t] = emb[(size_t)i * HID + t];
  __syncthreads();
  float acc = 0.f;
#pragma unroll 8
  for (int k = 0; k < HID; ++k) acc += er[k] * xa2[t * HID + k];
  out[(size_t)i * IN_DIM + t] = acc;
}

// -------- s_ = sigmoid(emb @ emb^T) : WMMA f16 path, one wave per 16x16 ------
__global__ void k_struct_dec(const _Float16* __restrict__ embh, float* __restrict__ s_out) {
  const int wave = threadIdx.x >> 5;
  const int lane = threadIdx.x & 31;
  const int tj = blockIdx.x * 8 + wave;        // column tile
  const int ti = blockIdx.y;                   // row tile
  if (tj >= NT) return;                        // wave-uniform -> EXEC all-ones inside

  // 16-bit A-fragment striping (ISA 7.12.2): lane&15 = row, lane>>4 = K-half
  const int r    = lane & 15;
  const int hsel = lane >> 4;
  const _Float16* arow = embh + (size_t)(ti * 16 + r) * HID + hsel * 8;
  const _Float16* brow = embh + (size_t)(tj * 16 + r) * HID + hsel * 8;  // B = emb^T

  v8f c = {};
#pragma unroll
  for (int kc = 0; kc < 2; ++kc) {             // K = 64 as 2 x 32
    const _Float16* ap = arow + kc * 32;
    const _Float16* bp = brow + kc * 32;
    v8h alo = *(const v8h*)ap;
    v8h ahi = *(const v8h*)(ap + 16);
    v8h blo = *(const v8h*)bp;
    v8h bhi = *(const v8h*)(bp + 16);
    v16h a, b;
#pragma unroll
    for (int q = 0; q < 8; ++q) {
      a[q] = alo[q]; a[q + 8] = ahi[q];
      b[q] = blo[q]; b[q + 8] = bhi[q];
    }
    c = __builtin_amdgcn_wmma_f32_16x16x32_f16(false, a, false, b, (short)0, c,
                                               false, false);
  }

  // C layout: VGPR v -> row v (lanes 0-15) / row v+8 (lanes 16-31), col = lane&15
  const int col   = tj * 16 + (lane & 15);
  const int rbase = ti * 16 + ((lane >> 4) ? 8 : 0);
#pragma unroll
  for (int v = 0; v < 8; ++v) {
    // sigmoid with 1-ulp hardware rcp (arg in (1,inf): safe) instead of the
    // IEEE-correct v_div_scale/newton chain; output is write-once -> NT store
    float val = __builtin_amdgcn_rcpf(1.0f + __expf(-c[v]));
    __builtin_nontemporal_store(val, &s_out[(size_t)(rbase + v) * N_NODES + col]);
  }
}

// ----------------------------------------------------------------------------
extern "C" void kernel_launch(void* const* d_in, const int* in_sizes, int n_in,
                              void* d_out, int out_size, void* d_ws, size_t ws_size,
                              hipStream_t stream) {
  (void)n_in; (void)out_size; (void)ws_size;
  const float* x       = (const float*)d_in[0];
  const int*   ei      = (const int*)  d_in[1];
  const float* W_stru  = (const float*)d_in[3];
  const float* b_stru  = (const float*)d_in[4];
  const float* W_gat   = (const float*)d_in[5];
  const float* att_src = (const float*)d_in[6];
  const float* att_dst = (const float*)d_in[7];
  const float* b_gat   = (const float*)d_in[8];
  const float* W_attr1 = (const float*)d_in[9];
  const float* b_attr1 = (const float*)d_in[10];
  const float* W_attr2 = (const float*)d_in[11];
  const float* b_attr2 = (const float*)d_in[12];
  const int E = in_sizes[1] / 2;

  // workspace layout (256B aligned slices)
  char* ws = (char*)d_ws;
  size_t o = 0;
  auto nxt = [&](size_t bytes) {
    size_t r = o;
    o += (bytes + 255) & ~(size_t)255;
    return r;
  };
  float*     h_buf   = (float*)(ws + nxt((size_t)N_NODES * EMB * 4));
  float*     hp_buf  = (float*)(ws + nxt((size_t)N_NODES * HID * 4));
  float*     a_src_b = (float*)(ws + nxt((size_t)N_NODES * 4));
  float*     a_dst_b = (float*)(ws + nxt((size_t)N_NODES * 4));
  float*     emb_b   = (float*)(ws + nxt((size_t)N_NODES * HID * 4));
  _Float16*  embh_b  = (_Float16*)(ws + nxt((size_t)N_NODES * HID * 2));
  float*     xa_b    = (float*)(ws + nxt((size_t)IN_DIM * EMB * 4));
  float*     xa2_b   = (float*)(ws + nxt((size_t)IN_DIM * HID * 4));
  int*       indeg   = (int*)(ws + nxt((size_t)N_NODES * 4));
  int*       offsets = (int*)(ws + nxt((size_t)(N_NODES + 1) * 4));
  int*       cursor  = (int*)(ws + nxt((size_t)N_NODES * 4));
  int*       csr_src = (int*)(ws + nxt((size_t)(E + N_NODES) * 4));

  float* out_x = (float*)d_out;                                  // [N, IN_DIM]
  float* out_s = (float*)d_out + (size_t)N_NODES * IN_DIM;       // [N, N]

  // encoders
  k_h <<<N_NODES, 64, 0, stream>>>(x, W_stru, b_stru, h_buf);
  k_hp<<<N_NODES, 64, 0, stream>>>(h_buf, W_gat, att_src, att_dst,
                                   hp_buf, a_src_b, a_dst_b);
  // CSR build (dst-grouped, incl. self loops)
  k_zero_i32<<<(N_NODES + 255) / 256, 256, 0, stream>>>(indeg, N_NODES);
  k_count<<<(E + N_NODES + 255) / 256, 256, 0, stream>>>(ei, E, indeg);
  k_scan<<<1, 1024, 0, stream>>>(indeg, offsets, cursor);
  k_fill<<<(E + N_NODES + 255) / 256, 256, 0, stream>>>(ei, E, cursor, csr_src);
  // GAT aggregation
  k_gat<<<N_NODES, 64, 0, stream>>>(offsets, csr_src, a_src_b, a_dst_b,
                                    hp_buf, b_gat, emb_b, embh_b);
  // attribute path + attribute decoder
  k_attr1<<<IN_DIM, 64, 0, stream>>>(x, W_attr1, b_attr1, xa_b);
  k_attr2<<<IN_DIM, 64, 0, stream>>>(xa_b, W_attr2, b_attr2, xa2_b);
  k_xrec <<<N_NODES, 128, 0, stream>>>(emb_b, xa2_b, out_x);
  // structure decoder (the bandwidth-bound 400 MB stage, WMMA compute)
  dim3 gs((NT + 7) / 8, NT);
  k_struct_dec<<<gs, 256, 0, stream>>>(embh_b, out_s);
}
